// CausalSelfAttention_67894843015857
// MI455X (gfx1250) — matrix operand order repaired and checked by reading, not compiled, not run on hardware
//
#include <hip/hip_runtime.h>

// ---------------- problem constants (match reference) ----------------
constexpr int cB = 2;
constexpr int cT = 2048;
constexpr int cC = 1024;        // N_EMBD
constexpr int cH = 16;          // N_HEAD
constexpr int cHD = 64;         // head dim
constexpr int cBLOCK = 2048;
constexpr int c3C = 3 * cC;

// ---------------- types ----------------
typedef __attribute__((ext_vector_type(16))) __bf16 v16bf;
typedef __attribute__((ext_vector_type(8)))  float  v8f;
typedef __attribute__((ext_vector_type(4)))  unsigned int v4u;

union FragU { v4u u[2]; v16bf v; };

static __device__ __forceinline__ __bf16 f2bf(float f) {
  union { float f; unsigned u; } c; c.f = f;
  unsigned r = c.u + 0x7FFFu + ((c.u >> 16) & 1u);   // round-to-nearest-even
  union { unsigned short s; __bf16 b; } o; o.s = (unsigned short)(r >> 16);
  return o.b;
}

// A-matrix fragment (16x32 bf16, M x K), source row-major, ld in elements (mult of 8).
// lane l: m = l&15, K-half = l>>4 ; VGPR0-3 <- K kh*8..+7 ; VGPR4-7 <- K 16+kh*8..+7
static __device__ __forceinline__ v16bf load_a_frag(const __bf16* base, int ld) {
  unsigned lane = threadIdx.x & 31u;
  unsigned m = lane & 15u, kh = lane >> 4;
  const __bf16* p = base + (size_t)m * ld + kh * 8;
  FragU f;
  f.u[0] = *(const v4u*)(p);
  f.u[1] = *(const v4u*)(p + 16);
  return f.v;
}

// B-matrix fragment (32x16 bf16, K x N) from an n-major (transposed) buffer:
// element [n][k] contiguous in k, row stride ld (mult of 8).
// lane l: n = l&15, K-half = l>>4 ; VGPR0-3 <- K kh*8..+7 ; VGPR4-7 <- K 16+kh*8..+7
static __device__ __forceinline__ v16bf load_bt_frag(const __bf16* base, int ld) {
  unsigned lane = threadIdx.x & 31u;
  unsigned n = lane & 15u, kh = lane >> 4;
  const __bf16* p = base + (size_t)n * ld + kh * 8;
  FragU f;
  f.u[0] = *(const v4u*)(p);
  f.u[1] = *(const v4u*)(p + 16);
  return f.v;
}

static __device__ __forceinline__ v8f wmma_bf16(v16bf a, v16bf b, v8f c) {
  return __builtin_amdgcn_wmma_f32_16x16x32_bf16(false, a, false, b, (short)0, c,
                                                 false, false);
}

// ============================================================================
// Kernel 1: per-batch counts + inclusive scan of 1/count, and gathered padding
// ============================================================================
__global__ __launch_bounds__(1024) void pos_kernel(const int* __restrict__ tok,
                                                   const float* __restrict__ pmask,
                                                   float* __restrict__ partial,
                                                   float* __restrict__ pmg) {
  __shared__ float cnt[cBLOCK];
  __shared__ float buf0[cT];
  __shared__ float buf1[cT];
  const int b = blockIdx.x;
  const int tid = threadIdx.x;
  for (int i = tid; i < cBLOCK; i += 1024) cnt[i] = 0.f;
  __syncthreads();
  for (int i = tid; i < cT; i += 1024) {
    atomicAdd(&cnt[tok[b * cT + i]], 1.0f);
  }
  __syncthreads();
  for (int i = tid; i < cT; i += 1024) {
    int idx = tok[b * cT + i];
    buf0[i] = 1.0f / cnt[idx];
    pmg[b * cT + i] = pmask[b * cBLOCK + idx];
  }
  __syncthreads();
  float* src = buf0; float* dst = buf1;
  for (int off = 1; off < cT; off <<= 1) {
    for (int i = tid; i < cT; i += 1024) {
      float v = src[i];
      if (i >= off) v += src[i - off];
      dst[i] = v;
    }
    __syncthreads();
    float* t = src; src = dst; dst = t;
  }
  for (int i = tid; i < cT; i += 1024) partial[b * cT + i] = src[i];
}

// ============================================================================
// Kernel 2/5: C = A(f32,MxK) * B(f32,KxN) + bias, bf16 WMMA, 128x64 block tile,
// double-buffered LDS pipeline (global loads overlap WMMA via split counters)
// ============================================================================
#define GBM 128
#define GBN 64
#define GKT 32
#define GLDA 40   // padded LDS stride (multiple of 8 -> 16B aligned frag reads)

__global__ __launch_bounds__(256) void gemm_bias_kernel(const float* __restrict__ A,
                                                        const float* __restrict__ Bw,
                                                        const float* __restrict__ bias,
                                                        float* __restrict__ Out,
                                                        int M, int N, int K) {
  __shared__ __align__(16) __bf16 As[2][GBM * GLDA];
  __shared__ __align__(16) __bf16 Bt[2][GBN * GLDA];
  const int tid = threadIdx.x;
  const int wave = tid >> 5;
  const unsigned lane = tid & 31u;
  const unsigned nl = lane & 15u, kh = lane >> 4;
  const int m0 = blockIdx.y * GBM;
  const int n0 = blockIdx.x * GBN;
  const int wi = wave >> 1;   // 0..3  (row tile of 32)
  const int wj = wave & 1;    // 0..1  (col tile of 32)

  // staging coordinates (fixed per thread)
  const int kk = tid >> 3;               // 0..31  (B k-row)
  const int nn = (tid & 7) << 3;         // 0..56  (B n-col base)

  float4 avr[4];                         // A staging registers
  float4 bvr[2];                         // B staging registers

  auto load_regs = [&](int kq) {
#pragma unroll
    for (int i = 0; i < 4; ++i) {
      int e = tid + i * 256;             // float4 index 0..1023
      int r = e >> 3;
      int c4 = (e & 7) << 2;
      avr[i] = *(const float4*)(A + (size_t)(m0 + r) * K + kq + c4);
    }
    const float* bp = Bw + (size_t)(kq + kk) * N + n0 + nn;
    bvr[0] = *(const float4*)(bp);
    bvr[1] = *(const float4*)(bp + 4);
  };

  auto store_stage = [&](int s) {
#pragma unroll
    for (int i = 0; i < 4; ++i) {
      int e = tid + i * 256;
      int r = e >> 3;
      int c4 = (e & 7) << 2;
      __bf16* d = &As[s][r * GLDA + c4];
      d[0] = f2bf(avr[i].x); d[1] = f2bf(avr[i].y);
      d[2] = f2bf(avr[i].z); d[3] = f2bf(avr[i].w);
    }
    __bf16* bd = Bt[s];
    bd[(nn + 0) * GLDA + kk] = f2bf(bvr[0].x);
    bd[(nn + 1) * GLDA + kk] = f2bf(bvr[0].y);
    bd[(nn + 2) * GLDA + kk] = f2bf(bvr[0].z);
    bd[(nn + 3) * GLDA + kk] = f2bf(bvr[0].w);
    bd[(nn + 4) * GLDA + kk] = f2bf(bvr[1].x);
    bd[(nn + 5) * GLDA + kk] = f2bf(bvr[1].y);
    bd[(nn + 6) * GLDA + kk] = f2bf(bvr[1].z);
    bd[(nn + 7) * GLDA + kk] = f2bf(bvr[1].w);
  };

  v8f acc00 = {}, acc01 = {}, acc10 = {}, acc11 = {};

  // prologue: stage tile 0
  load_regs(0);
  store_stage(0);
  __syncthreads();

  int buf = 0;
  for (int k0 = 0; k0 < K; k0 += GKT) {
    const int kn = k0 + GKT;
    // issue next tile's global loads early (LOADcnt), independent of LDS reads
    if (kn < K) {
      load_regs(kn);
      // L2 prefetch two tiles ahead (global_prefetch_b8)
      if (kn + GKT < K && tid < 64)
        __builtin_prefetch(A + (size_t)(m0 + tid * 2) * K + kn + GKT, 0, 1);
    }

    // compute on current buffer (DScnt path only)
    v16bf am0 = load_a_frag(&As[buf][(wi * 32 + 0)  * GLDA], GLDA);
    v16bf am1 = load_a_frag(&As[buf][(wi * 32 + 16) * GLDA], GLDA);
    v16bf bn0 = load_bt_frag(&Bt[buf][(wj * 32 + 0)  * GLDA], GLDA);
    v16bf bn1 = load_bt_frag(&Bt[buf][(wj * 32 + 16) * GLDA], GLDA);

    acc00 = wmma_bf16(am0, bn0, acc00);
    acc01 = wmma_bf16(am0, bn1, acc01);
    acc10 = wmma_bf16(am1, bn0, acc10);
    acc11 = wmma_bf16(am1, bn1, acc11);

    // convert + stage next tile after WMMAs (s_wait_loadcnt lands here)
    if (kn < K) store_stage(buf ^ 1);
    __syncthreads();
    buf ^= 1;
  }

  const int rbase = m0 + wi * 32;
  const int cbase = n0 + wj * 32;
  for (int r = 0; r < 8; ++r) {
    int row0 = rbase + r + 8 * (int)kh;
    int row1 = row0 + 16;
    int col0 = cbase + (int)nl;
    int col1 = col0 + 16;
    Out[(size_t)row0 * N + col0] = acc00[r] + bias[col0];
    Out[(size_t)row0 * N + col1] = acc01[r] + bias[col1];
    Out[(size_t)row1 * N + col0] = acc10[r] + bias[col0];
    Out[(size_t)row1 * N + col1] = acc11[r] + bias[col1];
  }
}

// ============================================================================
// Kernel 3: RoPE + fork + v-scale; emit bf16 q[B,H,T,64], k[B,H,T,64], vT[B,H,64,T]
// ============================================================================
__global__ __launch_bounds__(256) void rope_kernel(const float* __restrict__ qkv,
                                                   const float* __restrict__ partial,
                                                   const float* __restrict__ cs,
                                                   __bf16* __restrict__ qb,
                                                   __bf16* __restrict__ kb,
                                                   __bf16* __restrict__ vT) {
  int gid = blockIdx.x * blockDim.x + threadIdx.x;
  int j = gid & 63;
  int t = (gid >> 6) & (cT - 1);
  int h = (gid >> 17) & (cH - 1);
  int b = gid >> 21;
  if (b >= cB) return;

  size_t base = ((size_t)b * cT + t) * c3C + h * cHD;
  float pos = partial[b * cT + t];
  int i = j & 31;
  float inv = __powf(10000.0f, -(float)(2 * i) / 64.0f);
  float ang = pos * inv;
  float co = __cosf(ang), si = __sinf(ang);

  float qj = qkv[base + j];
  float kj = qkv[base + cC + j];
  float vj = qkv[base + 2 * cC + j];
  int pj = (j < 32) ? j + 32 : j - 32;
  float qp = qkv[base + pj];
  float kp = qkv[base + cC + pj];
  float rq = (j < 32) ? -qp : qp;
  float rk = (j < 32) ? -kp : kp;

  float qo = qj * co + rq * si;
  float ko = kj * co + rk * si;
  float csv = cs[b * cT + t];
  if (j == 63) { qo = 1.0f; ko = csv; }          // fork plan
  float vo = vj * __expf(csv);

  size_t hoff = (((size_t)(b * cH + h) * cT) + t) * cHD + j;
  qb[hoff] = f2bf(qo);
  kb[hoff] = f2bf(ko);
  vT[((size_t)(b * cH + h) * cHD + j) * cT + t] = f2bf(vo);
}

// ============================================================================
// Kernel 4: flash attention, 1 wave = 16 q rows, chunks of 32 keys, WMMA both GEMMs
// ============================================================================
__global__ __launch_bounds__(128) void attn_kernel(const __bf16* __restrict__ qb,
                                                   const __bf16* __restrict__ kb,
                                                   const __bf16* __restrict__ vT,
                                                   const float* __restrict__ pmg,
                                                   float* __restrict__ y) {
  __shared__ __align__(16) __bf16 ptile[4][16 * 32];   // per-wave P staging
  const int b = blockIdx.z, h = blockIdx.y;
  const int wave = threadIdx.x >> 5;
  const unsigned lane = threadIdx.x & 31u;
  const unsigned nl = lane & 15u, kh = lane >> 4;
  const int qbase = blockIdx.x * 64 + wave * 16;

  const __bf16* qhb = qb + ((size_t)(b * cH + h) * cT + qbase) * cHD;
  const __bf16* khb = kb + ((size_t)(b * cH + h) * cT) * cHD;
  const __bf16* vhb = vT + ((size_t)(b * cH + h) * cHD) * cT;

  v16bf a0 = load_a_frag(qhb, cHD);           // d = 0..31
  v16bf a1 = load_a_frag(qhb + 32, cHD);      // d = 32..63

  v8f acc0 = {}, acc1 = {}, acc2 = {}, acc3 = {};
  float mrow[8], lrow[8], pmq[8];
  for (int r = 0; r < 8; ++r) {
    mrow[r] = -1e30f; lrow[r] = 0.f;
    pmq[r] = pmg[b * cT + qbase + r + 8 * (int)kh];
  }
  const float scale = 0.125f;                 // 1/sqrt(64)
  const int kend = qbase + 16;                // causal limit (exclusive)

  for (int kj = 0; kj < kend; kj += 32) {
    // ---- S = Q K^T (two 16-col tiles) ----
    v16bf b00 = load_bt_frag(khb + (size_t)kj * cHD, cHD);
    v16bf b01 = load_bt_frag(khb + (size_t)kj * cHD + 32, cHD);
    v16bf b10 = load_bt_frag(khb + (size_t)(kj + 16) * cHD, cHD);
    v16bf b11 = load_bt_frag(khb + (size_t)(kj + 16) * cHD + 32, cHD);
    v8f s0 = {}, s1 = {};
    s0 = wmma_bf16(a0, b00, s0);  s0 = wmma_bf16(a1, b01, s0);
    s1 = wmma_bf16(a0, b10, s1);  s1 = wmma_bf16(a1, b11, s1);

    // issue V fragment loads early: overlap with softmax VALU/trans work
    v16bf bv0 = load_bt_frag(vhb + (size_t)0  * cT + kj, cT);
    v16bf bv1 = load_bt_frag(vhb + (size_t)16 * cT + kj, cT);
    v16bf bv2 = load_bt_frag(vhb + (size_t)32 * cT + kj, cT);
    v16bf bv3 = load_bt_frag(vhb + (size_t)48 * cT + kj, cT);

    const int col0 = kj + (int)nl;
    const int col1 = kj + 16 + (int)nl;
    const float pmc0 = pmg[b * cT + col0];
    const float pmc1 = pmg[b * cT + col1];

    // ---- scale + causal/padding mask ----
    for (int r = 0; r < 8; ++r) {
      int row = qbase + r + 8 * (int)kh;
      float e0 = s0[r] * scale;
      float e1 = s1[r] * scale;
      if (col0 > row || pmq[r] * pmc0 == 0.f) e0 = -1e30f;
      if (col1 > row || pmq[r] * pmc1 == 0.f) e1 = -1e30f;
      s0[r] = e0; s1[r] = e1;
    }

    // ---- online softmax (row reductions within 16-lane halves) ----
    for (int r = 0; r < 8; ++r) {
      float cm = fmaxf(s0[r], s1[r]);
      cm = fmaxf(cm, __shfl_xor(cm, 1, 32));
      cm = fmaxf(cm, __shfl_xor(cm, 2, 32));
      cm = fmaxf(cm, __shfl_xor(cm, 4, 32));
      cm = fmaxf(cm, __shfl_xor(cm, 8, 32));
      float mnew = fmaxf(mrow[r], cm);
      float alpha = __expf(mrow[r] - mnew);
      float p0 = __expf(s0[r] - mnew);
      float p1 = __expf(s1[r] - mnew);
      float rs = p0 + p1;
      rs += __shfl_xor(rs, 1, 32);
      rs += __shfl_xor(rs, 2, 32);
      rs += __shfl_xor(rs, 4, 32);
      rs += __shfl_xor(rs, 8, 32);
      lrow[r] = lrow[r] * alpha + rs;
      mrow[r] = mnew;
      s0[r] = p0; s1[r] = p1;
      acc0[r] *= alpha; acc1[r] *= alpha; acc2[r] *= alpha; acc3[r] *= alpha;
    }

    // ---- C-layout P -> LDS -> A-layout fragment (wave-private, dscnt only) ----
    __bf16* pt = ptile[wave];
    for (int r = 0; r < 8; ++r) {
      pt[(r + 8 * kh) * 32 + nl]      = f2bf(s0[r]);
      pt[(r + 8 * kh) * 32 + 16 + nl] = f2bf(s1[r]);
    }
    asm volatile("s_wait_dscnt 0" ::: "memory");
    v16bf pa = load_a_frag(pt, 32);

    // ---- O += P V ----
    acc0 = wmma_bf16(pa, bv0, acc0);
    acc1 = wmma_bf16(pa, bv1, acc1);
    acc2 = wmma_bf16(pa, bv2, acc2);
    acc3 = wmma_bf16(pa, bv3, acc3);
  }

  // ---- normalize and store y[b, t, h*64 + d] (f32, GEMM-ready) ----
  for (int r = 0; r < 8; ++r) {
    int row = qbase + r + 8 * (int)kh;
    float invl = 1.0f / lrow[r];
    size_t obase = ((size_t)b * cT + row) * cC + h * cHD;
    y[obase + 0  + nl] = acc0[r] * invl;
    y[obase + 16 + nl] = acc1[r] * invl;
    y[obase + 32 + nl] = acc2[r] * invl;
    y[obase + 48 + nl] = acc3[r] * invl;
  }
}

// ============================================================================
// launch
// ============================================================================
extern "C" void kernel_launch(void* const* d_in, const int* in_sizes, int n_in,
                              void* d_out, int out_size, void* d_ws, size_t ws_size,
                              hipStream_t stream) {
  const float* x     = (const float*)d_in[0];
  const float* cs    = (const float*)d_in[1];
  const float* pmask = (const float*)d_in[2];
  const float* Wa    = (const float*)d_in[3];
  const float* ba    = (const float*)d_in[4];
  const float* Wp    = (const float*)d_in[5];
  const float* bp    = (const float*)d_in[6];
  const int*   tok   = (const int*)d_in[7];

  char* ws = (char*)d_ws;
  float* partial = (float*)ws;  ws += (size_t)cB * cT * sizeof(float);
  float* pmg     = (float*)ws;  ws += (size_t)cB * cT * sizeof(float);
  float* qkv     = (float*)ws;  ws += (size_t)cB * cT * c3C * sizeof(float);
  __bf16* qb     = (__bf16*)ws; ws += (size_t)cB * cH * cT * cHD * sizeof(__bf16);
  __bf16* kb     = (__bf16*)ws; ws += (size_t)cB * cH * cT * cHD * sizeof(__bf16);
  __bf16* vT     = (__bf16*)ws; ws += (size_t)cB * cH * cT * cHD * sizeof(__bf16);
  float* yb      = (float*)ws;

  const int M = cB * cT;  // 4096

  pos_kernel<<<dim3(cB), dim3(1024), 0, stream>>>(tok, pmask, partial, pmg);

  gemm_bias_kernel<<<dim3(c3C / GBN, M / GBM), dim3(256), 0, stream>>>(
      x, Wa, ba, qkv, M, c3C, cC);

  rope_kernel<<<dim3((cB * cH * cT * cHD) / 256), dim3(256), 0, stream>>>(
      qkv, partial, cs, qb, kb, vT);

  attn_kernel<<<dim3(cT / 64, cH, cB), dim3(128), 0, stream>>>(qb, kb, vT, pmg, yb);

  gemm_bias_kernel<<<dim3(cC / GBN, M / GBM), dim3(256), 0, stream>>>(
      yb, Wp, bp, (float*)d_out, M, cC, cC);
}
